// OcrLoss_67001489817773
// MI455X (gfx1250) — matrix operand order repaired and checked by reading, not compiled, not run on hardware
//
#include <hip/hip_runtime.h>
#include <stdint.h>

typedef __attribute__((ext_vector_type(16))) _Float16 v16h;
typedef __attribute__((ext_vector_type(8)))  float    v8f;

union AFrag { v16h v; _Float16 h[16]; uint32_t u[8]; uint4 q[2]; };

__device__ __forceinline__ float sigmoidf_(float x) {
    return 1.0f / (1.0f + __expf(-x));
}

// ---------------------------------------------------------------------------
// 1. ROI max pool: base_feat [8,512,40,160] + rois [32,5] -> rnn_in f16
//    [64,32,1024] (row = t*32+n, col = c*2+ph). Matches reference roi_pool.
// ---------------------------------------------------------------------------
__global__ __launch_bounds__(128) void ocr_roi_pool_kernel(
    const float* __restrict__ feat, const float* __restrict__ rois,
    _Float16* __restrict__ rnn16)
{
    const int c  = blockIdx.x;          // 0..511
    const int n  = blockIdx.y;          // 0..31
    const int t  = threadIdx.x;         // 0..127
    const int ph = t >> 6, pw = t & 63;

    const float* r = rois + n * 5;
    const int bi = (int)r[0];
    const int x1 = (int)rintf(r[1] * 160.0f);
    const int y1 = (int)rintf(r[2] * 40.0f);
    const int x2 = (int)rintf(r[3] * 160.0f);
    const int y2 = (int)rintf(r[4] * 40.0f);
    const float rw = (float)max(x2 - x1 + 1, 1);
    const float rh = (float)max(y2 - y1 + 1, 1);
    const float bh = rh * 0.5f;            // / POOLED_H
    const float bw = rw * (1.0f / 64.0f);  // / POOLED_W

    int hs = min(max((int)floorf((float)ph * bh) + y1, 0), 40);
    int he = min(max((int)ceilf((float)(ph + 1) * bh) + y1, 0), 40);
    int ws = min(max((int)floorf((float)pw * bw) + x1, 0), 160);
    int we = min(max((int)ceilf((float)(pw + 1) * bw) + x1, 0), 160);

    const float* f = feat + ((size_t)bi * 512 + c) * (40 * 160);
    float m = -1e30f;
    bool any = false;
    for (int h = hs; h < he; ++h)
        for (int w = ws; w < we; ++w) {
            any = true;
            m = fmaxf(m, f[h * 160 + w]);
        }
    const float out = any ? m : 0.0f;

    rnn16[((size_t)pw * 32 + n) * 1024 + (size_t)c * 2 + ph] = (_Float16)out;
}

// ---------------------------------------------------------------------------
// 2. Pack W (f32 [Nrows,K]) as B = W^T in WMMA B-fragment order (f16):
//    Bp[(((ntile*KC)+kc)*32 + lane)*8 + r] = pack(W[n,k], W[n,k+1])
//    n = ntile*16 + lane%16, k = kc*32 + (r<4 ? (lane/16)*8 + 2r : 16 + (lane/16)*8 + 2(r-4)), KC = K/32.
//    One fragment load = two lane-contiguous global_load_b128.
// ---------------------------------------------------------------------------
__global__ __launch_bounds__(256) void ocr_pack_b_kernel(
    const float* __restrict__ W, uint32_t* __restrict__ Bp,
    int K, int N, int Nvalid)
{
    const int KC = K >> 5;
    const int total = (N >> 4) * KC * 256;
    int id = blockIdx.x * 256 + threadIdx.x;
    if (id >= total) return;
    const int r    = id & 7;
    const int lane = (id >> 3) & 31;
    const int rest = id >> 8;
    const int kc    = rest % KC;
    const int ntile = rest / KC;
    const int g = lane >> 4, nl = lane & 15;
    const int k = kc * 32 + (r < 4 ? g * 8 + 2 * r : 16 + g * 8 + 2 * (r - 4));
    const int n = ntile * 16 + nl;
    union { _Float16 h[2]; uint32_t u; } p;
    const bool ok = (n < Nvalid);
    p.h[0] = (_Float16)(ok ? W[(size_t)n * K + k] : 0.0f);
    p.h[1] = (_Float16)(ok ? W[(size_t)n * K + k + 1] : 0.0f);
    Bp[id] = p.u;
}

__global__ __launch_bounds__(256) void ocr_bias_sum_kernel(
    const float* __restrict__ a, const float* __restrict__ b,
    float* __restrict__ o, int n)
{
    int i = blockIdx.x * 256 + threadIdx.x;
    if (i < n) o[i] = a[i] + b[i];
}

__global__ __launch_bounds__(256) void ocr_pad_bias_kernel(
    const float* __restrict__ src, float* __restrict__ dst, int nvalid, int n)
{
    int i = blockIdx.x * 256 + threadIdx.x;
    if (i < n) dst[i] = (i < nvalid) ? src[i] : 0.0f;
}

// ---------------------------------------------------------------------------
// 3. WMMA GEMM, no LDS: C[M,N] = A[M,K](f16 row-major) * Bpack + bias.
//    Block 256 thr = 8 waves; wave tile 32x64 (2 m-tiles x 4 n-tiles);
//    block tile 256x64. All reuse via L2 (192 MB).
//    Epilogue: row-major f32 (C), row-major f16 (C16), or f32 fragment
//    layout (Cfrag: [N/16][M/16][32][8]) for direct LSTM consumption.
// ---------------------------------------------------------------------------
__global__ __launch_bounds__(256) void ocr_gemm_f16_kernel(
    const _Float16* __restrict__ A, const uint32_t* __restrict__ Bp,
    const float* __restrict__ bias,
    float* __restrict__ C, _Float16* __restrict__ C16,
    float* __restrict__ Cfrag,
    int M, int N, int K)
{
    const int tid  = threadIdx.x;
    const int wv   = tid >> 5;
    const int lane = tid & 31;
    const int g    = lane >> 4;
    const int nl   = lane & 15;
    const int KC   = K >> 5;
    const int MT   = M >> 4;
    const int nbase  = blockIdx.x * 64;
    const int ntile0 = nbase >> 4;
    const int mtile0 = blockIdx.y * 16 + wv * 2;   // wave's first m-tile

    v8f acc[2][4];
    const v8f zero = {0.f, 0.f, 0.f, 0.f, 0.f, 0.f, 0.f, 0.f};
#pragma unroll
    for (int mt = 0; mt < 2; ++mt)
#pragma unroll
        for (int s = 0; s < 4; ++s) acc[mt][s] = zero;

    for (int kc = 0; kc < KC; ++kc) {
        AFrag af[2];
#pragma unroll
        for (int mt = 0; mt < 2; ++mt) {
            const _Float16* arow =
                A + (size_t)((mtile0 + mt) * 16 + nl) * K + kc * 32 + g * 8;
            af[mt].q[0] = *(const uint4*)arow;
            af[mt].q[1] = *(const uint4*)(arow + 16);
        }
#pragma unroll
        for (int s = 0; s < 4; ++s) {
            AFrag bf;
            const uint4* p = (const uint4*)(
                Bp + ((size_t)((ntile0 + s) * KC + kc) * 32 + lane) * 8);
            bf.q[0] = p[0];
            bf.q[1] = p[1];
#pragma unroll
            for (int mt = 0; mt < 2; ++mt)
                acc[mt][s] = __builtin_amdgcn_wmma_f32_16x16x32_f16(
                    false, af[mt].v, false, bf.v, (short)0, acc[mt][s],
                    false, false);
        }
    }

#pragma unroll
    for (int s = 0; s < 4; ++s) {
        const int col = nbase + s * 16 + nl;
        const float bv = bias ? bias[col] : 0.0f;
#pragma unroll
        for (int mt = 0; mt < 2; ++mt) {
            v8f val = acc[mt][s];
#pragma unroll
            for (int v = 0; v < 8; ++v) val[v] += bv;
            if (Cfrag) {
                float* fp = Cfrag +
                    (((size_t)(ntile0 + s) * MT + (mtile0 + mt)) * 32 + lane) * 8;
                *(v8f*)fp = val;
            }
            if (C || C16) {
#pragma unroll
                for (int v = 0; v < 8; ++v) {
                    const int row = (mtile0 + mt) * 16 + v + 8 * g;
                    if (C)   C[(size_t)row * N + col]   = val[v];
                    if (C16) C16[(size_t)row * N + col] = (_Float16)val[v];
                }
            }
        }
    }
}

// ---------------------------------------------------------------------------
// 4. Persistent bidirectional LSTM scan. grid=2 (dir), 512 thr = 16 waves.
//    Wave w: batch-row tile mt = w&1, hidden cols j0 = (w>>1)*32.
//    xg is in f32 fragment layout [dir][64 ntiles][128 mtiles][32][8];
//    W_hh is fragment-packed; h state lives in LDS as f16.
// ---------------------------------------------------------------------------
__global__ __launch_bounds__(512) void ocr_lstm_scan_kernel(
    const float* __restrict__ xg,      // [2][64][128][32][8] fragment f32
    const uint32_t* __restrict__ Bp,   // [2][131072] packed W_hh fragments
    _Float16* __restrict__ hcat)       // [2048][512], dir d -> cols d*256..
{
    const int dir  = blockIdx.x;
    const int tid  = threadIdx.x;
    const int wv   = tid >> 5;
    const int lane = tid & 31;
    const int g    = lane >> 4;
    const int nl   = lane & 15;
    const int mt   = wv & 1;
    const int j0   = (wv >> 1) * 32;

    __shared__ _Float16 h16[32 * 264];   // h state, stride 264 halfs
    for (int i = tid; i < 32 * 264; i += 512) h16[i] = (_Float16)0.0f;
    __syncthreads();

    const float*    xf = xg + (size_t)dir * 2048 * 1024;
    const uint32_t* bp = Bp + (size_t)dir * 131072;

    v8f acc[4][2];     // [gate i/f/g/o][n-subtile]
    v8f cst[2];        // cell state fragments
    const v8f zero = {0.f, 0.f, 0.f, 0.f, 0.f, 0.f, 0.f, 0.f};
    cst[0] = zero; cst[1] = zero;

    for (int tt = 0; tt < 64; ++tt) {
        const int t = dir ? (63 - tt) : tt;
        const int mtileG = t * 2 + mt;

        // init accumulators with xg[t] fragments (contain b_ih + b_hh)
#pragma unroll
        for (int gi = 0; gi < 4; ++gi)
#pragma unroll
            for (int s = 0; s < 2; ++s) {
                const int ntileG = gi * 16 + (wv >> 1) * 2 + s;
                acc[gi][s] = *(const v8f*)(
                    xf + (((size_t)ntileG * 128 + mtileG) * 32 + lane) * 8);
            }

        // gates += h · W_hh^T   (K = 256 in 8 chunks of 32)
#pragma unroll
        for (int kc = 0; kc < 8; ++kc) {
            AFrag af;
            {
                const _Float16* hrow = &h16[(mt * 16 + nl) * 264 + kc * 32 + g * 8];
                af.u[0] = *(const uint32_t*)(hrow + 0);
                af.u[1] = *(const uint32_t*)(hrow + 2);
                af.u[2] = *(const uint32_t*)(hrow + 4);
                af.u[3] = *(const uint32_t*)(hrow + 6);
                af.u[4] = *(const uint32_t*)(hrow + 16);
                af.u[5] = *(const uint32_t*)(hrow + 18);
                af.u[6] = *(const uint32_t*)(hrow + 20);
                af.u[7] = *(const uint32_t*)(hrow + 22);
            }
#pragma unroll
            for (int gi = 0; gi < 4; ++gi)
#pragma unroll
                for (int s = 0; s < 2; ++s) {
                    const int ntile = gi * 16 + (wv >> 1) * 2 + s;
                    AFrag bf;
                    const uint4* p = (const uint4*)(
                        bp + ((size_t)(ntile * 8 + kc) * 32 + lane) * 8);
                    bf.q[0] = p[0];
                    bf.q[1] = p[1];
                    acc[gi][s] = __builtin_amdgcn_wmma_f32_16x16x32_f16(
                        false, af.v, false, bf.v, (short)0, acc[gi][s],
                        false, false);
                }
        }
        __syncthreads();   // all waves done reading h16

        // elementwise LSTM cell, write new h (f16) to LDS + global
#pragma unroll
        for (int s = 0; s < 2; ++s)
#pragma unroll
            for (int v = 0; v < 8; ++v) {
                const float iv = sigmoidf_(acc[0][s][v]);
                const float fv = sigmoidf_(acc[1][s][v]);
                const float gv = tanhf(acc[2][s][v]);
                const float ov = sigmoidf_(acc[3][s][v]);
                const float c  = fv * cst[s][v] + iv * gv;
                cst[s][v] = c;
                const float h = ov * tanhf(c);
                const int row = mt * 16 + v + 8 * g;
                const int col = j0 + s * 16 + nl;
                const _Float16 hh = (_Float16)h;
                h16[row * 264 + col] = hh;
                hcat[(size_t)(t * 32 + row) * 512 + dir * 256 + col] = hh;
            }
        __syncthreads();   // h16 fully updated before next step
    }
}

// ---------------------------------------------------------------------------
// 5. CTC loss: logits [2048,64] (cols 0..11 valid), one wave, lane = batch.
// ---------------------------------------------------------------------------
__global__ __launch_bounds__(32) void ocr_ctc_kernel(
    const float* __restrict__ logits, const int* __restrict__ text,
    const int* __restrict__ tlen, float* __restrict__ out)
{
    const int b = threadIdx.x;      // 0..31
    const float NEG = -1e30f;

    const int tl = tlen[b];
    int off = 0;
    for (int i = 0; i < b; ++i) off += tlen[i];

    int ext[17];
#pragma unroll
    for (int s = 0; s < 17; ++s) ext[s] = 0;
#pragma unroll
    for (int l = 0; l < 8; ++l) {
        int idx = off + l;
        idx = idx < 0 ? 0 : (idx > 255 ? 255 : idx);
        ext[2 * l + 1] = (l < tl) ? text[idx] : 0;
    }
    bool skip[17];
#pragma unroll
    for (int s = 0; s < 17; ++s)
        skip[s] = (s >= 2) && (ext[s] != 0) && (ext[s] != ext[s - 2]);

    float a[17], lp[12];
    auto compute_lp = [&](int t) {
        const float* row = logits + (size_t)(t * 32 + b) * 64;
        float m = row[0];
#pragma unroll
        for (int k = 1; k < 12; ++k) m = fmaxf(m, row[k]);
        float s = 0.0f;
#pragma unroll
        for (int k = 0; k < 12; ++k) s += expf(row[k] - m);
        const float lse = m + logf(s);
#pragma unroll
        for (int k = 0; k < 12; ++k) lp[k] = row[k] - lse;
    };

    compute_lp(0);
#pragma unroll
    for (int s = 0; s < 17; ++s) a[s] = NEG;
    a[0] = lp[0];
    a[1] = (tl > 0) ? lp[ext[1]] : NEG;

    for (int t = 1; t < 64; ++t) {
        compute_lp(t);
        float na[17];
#pragma unroll
        for (int s = 0; s < 17; ++s) {
            const float a0 = a[s];
            const float a1 = (s >= 1) ? a[s - 1] : NEG;
            const float a2 = (s >= 2 && skip[s]) ? a[s - 2] : NEG;
            const float m  = fmaxf(a0, fmaxf(a1, a2));
            const float sm = expf(a0 - m) + expf(a1 - m) + expf(a2 - m);
            const float nv = m + logf(sm) + lp[ext[s]];
            na[s] = (s < 2 * tl + 1) ? nv : NEG;
        }
#pragma unroll
        for (int s = 0; s < 17; ++s) a[s] = na[s];
    }

    const int e1 = 2 * tl;
    const int e2 = (2 * tl - 1) < 0 ? 0 : (2 * tl - 1);
    const float aT1 = a[e1];
    const float aT2 = (tl > 0) ? a[e2] : NEG;
    const float m = fmaxf(aT1, aT2);
    float loss = -(m + logf(expf(aT1 - m) + expf(aT2 - m)));

    for (int o2 = 16; o2 > 0; o2 >>= 1) loss += __shfl_down(loss, o2);
    if (b == 0) out[0] = loss / 32.0f;
}

// ---------------------------------------------------------------------------
// Host launcher
// ---------------------------------------------------------------------------
extern "C" void kernel_launch(void* const* d_in, const int* in_sizes, int n_in,
                              void* d_out, int out_size, void* d_ws, size_t ws_size,
                              hipStream_t stream)
{
    (void)in_sizes; (void)n_in; (void)out_size; (void)ws_size;

    const float* base_feat = (const float*)d_in[0];
    const int*   text      = (const int*)d_in[1];
    const int*   tlen      = (const int*)d_in[2];
    const float* rois      = (const float*)d_in[3];
    // per (layer,dir): W_ih, W_hh, b_ih, b_hh at 4.., 8.., 12.., 16..
    const float* Wih1[2] = { (const float*)d_in[4],  (const float*)d_in[8]  };
    const float* Whh1[2] = { (const float*)d_in[5],  (const float*)d_in[9]  };
    const float* bih1[2] = { (const float*)d_in[6],  (const float*)d_in[10] };
    const float* bhh1[2] = { (const float*)d_in[7],  (const float*)d_in[11] };
    const float* Wih2[2] = { (const float*)d_in[12], (const float*)d_in[16] };
    const float* Whh2[2] = { (const float*)d_in[13], (const float*)d_in[17] };
    const float* bih2[2] = { (const float*)d_in[14], (const float*)d_in[18] };
    const float* bhh2[2] = { (const float*)d_in[15], (const float*)d_in[19] };
    const float* Wemb1 = (const float*)d_in[20];
    const float* bemb1 = (const float*)d_in[21];
    const float* Wemb2 = (const float*)d_in[22];
    const float* bemb2 = (const float*)d_in[23];

    // ---- workspace carve-out (all 256B aligned) ----
    size_t off = 0;
    auto alloc = [&](size_t bytes) -> void* {
        void* p = (char*)d_ws + off;
        off += (bytes + 255) & ~(size_t)255;
        return p;
    };
    _Float16*  rnn16 = (_Float16*)alloc((size_t)64 * 32 * 1024 * 2);
    uint32_t*  Bih1p[2] = { (uint32_t*)alloc((size_t)524288 * 4),
                            (uint32_t*)alloc((size_t)524288 * 4) };
    uint32_t*  Bih2p[2] = { (uint32_t*)alloc((size_t)131072 * 4),
                            (uint32_t*)alloc((size_t)131072 * 4) };
    uint32_t*  Bph1  = (uint32_t*)alloc((size_t)2 * 131072 * 4);
    uint32_t*  Bph2  = (uint32_t*)alloc((size_t)2 * 131072 * 4);
    uint32_t*  Bemb1 = (uint32_t*)alloc((size_t)65536 * 4);
    uint32_t*  Bemb2 = (uint32_t*)alloc((size_t)16384 * 4);
    float*     bsum1[2] = { (float*)alloc(1024 * 4), (float*)alloc(1024 * 4) };
    float*     bsum2[2] = { (float*)alloc(1024 * 4), (float*)alloc(1024 * 4) };
    float*     be2pad = (float*)   alloc(64 * 4);
    float*     xg     = (float*)   alloc((size_t)2 * 2048 * 1024 * 4);
    _Float16*  h1cat  = (_Float16*)alloc((size_t)2048 * 512 * 2);
    _Float16*  h1emb  = (_Float16*)alloc((size_t)2048 * 256 * 2);
    _Float16*  h2cat  = (_Float16*)alloc((size_t)2048 * 512 * 2);
    float*     logits = (float*)   alloc((size_t)2048 * 64 * 4);

    // ---- 1. ROI pool ----
    ocr_roi_pool_kernel<<<dim3(512, 32), 128, 0, stream>>>(base_feat, rois, rnn16);

    // ---- 2. weight prep (all B matrices fragment-packed) ----
    for (int d = 0; d < 2; ++d) {
        ocr_pack_b_kernel<<<524288 / 256, 256, 0, stream>>>(
            Wih1[d], Bih1p[d], 1024, 1024, 1024);
        ocr_pack_b_kernel<<<131072 / 256, 256, 0, stream>>>(
            Wih2[d], Bih2p[d], 256, 1024, 1024);
        ocr_pack_b_kernel<<<131072 / 256, 256, 0, stream>>>(
            Whh1[d], Bph1 + (size_t)d * 131072, 256, 1024, 1024);
        ocr_pack_b_kernel<<<131072 / 256, 256, 0, stream>>>(
            Whh2[d], Bph2 + (size_t)d * 131072, 256, 1024, 1024);
        ocr_bias_sum_kernel<<<4, 256, 0, stream>>>(bih1[d], bhh1[d], bsum1[d], 1024);
        ocr_bias_sum_kernel<<<4, 256, 0, stream>>>(bih2[d], bhh2[d], bsum2[d], 1024);
    }
    ocr_pack_b_kernel<<<65536 / 256, 256, 0, stream>>>(Wemb1, Bemb1, 512, 256, 256);
    ocr_pack_b_kernel<<<16384 / 256, 256, 0, stream>>>(Wemb2, Bemb2, 512, 64, 12);
    ocr_pad_bias_kernel<<<1, 256, 0, stream>>>(bemb2, be2pad, 12, 64);

    // ---- 3. layer-1 input gates: xg1[d] = rnn_in · W_ih1^T + b (frag f32) ----
    for (int d = 0; d < 2; ++d)
        ocr_gemm_f16_kernel<<<dim3(16, 8), 256, 0, stream>>>(
            rnn16, Bih1p[d], bsum1[d],
            nullptr, nullptr, xg + (size_t)d * 2048 * 1024, 2048, 1024, 1024);

    // ---- 4. layer-1 recurrent scan (fw + bw) ----
    ocr_lstm_scan_kernel<<<2, 512, 0, stream>>>(xg, Bph1, h1cat);

    // ---- 5. embed1: h1emb = h1cat · W_emb1^T + b_emb1 (row-major f16) ----
    ocr_gemm_f16_kernel<<<dim3(4, 8), 256, 0, stream>>>(
        h1cat, Bemb1, bemb1, nullptr, h1emb, nullptr, 2048, 256, 512);

    // ---- 6. layer-2 input gates (reuse xg buffer, frag f32) ----
    for (int d = 0; d < 2; ++d)
        ocr_gemm_f16_kernel<<<dim3(16, 8), 256, 0, stream>>>(
            h1emb, Bih2p[d], bsum2[d],
            nullptr, nullptr, xg + (size_t)d * 2048 * 1024, 2048, 1024, 256);

    // ---- 7. layer-2 recurrent scan ----
    ocr_lstm_scan_kernel<<<2, 512, 0, stream>>>(xg, Bph2, h2cat);

    // ---- 8. embed2: logits = h2cat · W_emb2^T + b_emb2 (N padded to 64) ----
    ocr_gemm_f16_kernel<<<dim3(1, 8), 256, 0, stream>>>(
        h2cat, Bemb2, be2pad, logits, nullptr, nullptr, 2048, 64, 512);

    // ---- 9. CTC loss ----
    ocr_ctc_kernel<<<1, 32, 0, stream>>>(logits, text, tlen, (float*)d_out);
}
